// SwitchPositionwiseFeedForward_7988639171108
// MI455X (gfx1250) — compile-verified, matrix operand order checked by reading, not run on hardware
//
#include <hip/hip_runtime.h>
#include <hip/hip_bf16.h>

#define TOK   65536
#define CDIM  256
#define HDIM  1024
#define NEXP  4
#define MT     48                // tokens per block tile (3 x 16)
#define MTILES 3
#define XS_STR (CDIM + 8)        // padded LDS strides (16B-aligned, conflict-free)
#define HS_STR (HDIM + 8)

typedef __attribute__((ext_vector_type(16))) __bf16 v16bf;
typedef __attribute__((ext_vector_type(8)))  __bf16 v8bf;
typedef __attribute__((ext_vector_type(8)))  float  v8f;

// Load one WMMA 16x16x32 bf16 A/B fragment (ISA 7.12.2 layout):
// lane holds elements [0..7] and [16..23] relative to its k0-adjusted base.
__device__ __forceinline__ v16bf load_frag(const __bf16* __restrict__ p) {
  union { v16bf v; v8bf h[2]; } u;
  u.h[0] = *(const v8bf*)(p);
  u.h[1] = *(const v8bf*)(p + 16);
  return u.v;
}

// ---------------- Kernel 0: weight convert + transpose ----------------
// W1 [E][C][H] f32 -> W1t [E][H][C] bf16
__global__ __launch_bounds__(256) void prep_w1_kernel(const float* __restrict__ W1,
                                                      __bf16* __restrict__ W1t) {
  size_t i = (size_t)blockIdx.x * 256 + threadIdx.x;   // over E*H*C
  int c = (int)(i & (CDIM - 1));
  int h = (int)((i >> 8) & (HDIM - 1));
  int e = (int)(i >> 18);
  W1t[i] = (__bf16)W1[((size_t)e * CDIM + c) * HDIM + h];
}
// W2 [E][H][C] f32 -> W2t [E][C][H] bf16
__global__ __launch_bounds__(256) void prep_w2_kernel(const float* __restrict__ W2,
                                                      __bf16* __restrict__ W2t) {
  size_t i = (size_t)blockIdx.x * 256 + threadIdx.x;   // over E*C*H
  int h = (int)(i & (HDIM - 1));
  int c = (int)((i >> 10) & (CDIM - 1));
  int e = (int)(i >> 18);
  W2t[i] = (__bf16)W2[((size_t)e * HDIM + h) * CDIM + c];
}

// ---------------- Kernel 1: router (softmax/argmax) + x->bf16 + bucketing ----------------
__global__ __launch_bounds__(256) void router_kernel(
    const float* __restrict__ x, const float* __restrict__ Wsw, const float* __restrict__ bsw,
    __bf16* __restrict__ xb, float* __restrict__ pmaxArr,
    int* __restrict__ bucket, int* __restrict__ cnt)
{
  const int wave = threadIdx.x >> 5;
  const int lane = threadIdx.x & 31;
  const int token = blockIdx.x * 8 + wave;     // grid = TOK/8 blocks, 8 waves/block

  const float* xr = x + (size_t)token * CDIM;
  float vals[8];
  {
    float4 a = *(const float4*)(xr + lane * 8);
    float4 b = *(const float4*)(xr + lane * 8 + 4);
    vals[0] = a.x; vals[1] = a.y; vals[2] = a.z; vals[3] = a.w;
    vals[4] = b.x; vals[5] = b.y; vals[6] = b.z; vals[7] = b.w;
  }
  float l0 = 0.f, l1 = 0.f, l2 = 0.f, l3 = 0.f;
#pragma unroll
  for (int i = 0; i < 8; ++i) {
    float4 w = *(const float4*)(Wsw + (size_t)(lane * 8 + i) * NEXP);
    l0 += vals[i] * w.x; l1 += vals[i] * w.y; l2 += vals[i] * w.z; l3 += vals[i] * w.w;
  }
#pragma unroll
  for (int off = 16; off >= 1; off >>= 1) {   // wave32 butterfly reduction
    l0 += __shfl_xor(l0, off);
    l1 += __shfl_xor(l1, off);
    l2 += __shfl_xor(l2, off);
    l3 += __shfl_xor(l3, off);
  }
  l0 += bsw[0]; l1 += bsw[1]; l2 += bsw[2]; l3 += bsw[3];

  int route = 0; float best = l0;              // first-max wins, matches jnp.argmax
  if (l1 > best) { best = l1; route = 1; }
  if (l2 > best) { best = l2; route = 2; }
  if (l3 > best) { best = l3; route = 3; }
  float s = __expf(l0 - best) + __expf(l1 - best) + __expf(l2 - best) + __expf(l3 - best);
  float pm = 1.0f / s;                          // max softmax prob

  if (lane == 0) {
    pmaxArr[token] = pm;
    int slot = atomicAdd(&cnt[route], 1);
    bucket[(size_t)route * TOK + slot] = token;
  }
  v8bf ob;
#pragma unroll
  for (int i = 0; i < 8; ++i) ob[i] = (__bf16)vals[i];
  *(v8bf*)(xb + (size_t)token * CDIM + lane * 8) = ob;
}

// ---------------- Kernel 2: per-expert 2-layer MLP via WMMA ----------------
__global__ __launch_bounds__(256) void moe_ffn_kernel(
    const __bf16* __restrict__ xb,
    const __bf16* __restrict__ W1t, const float* __restrict__ b1,
    const __bf16* __restrict__ W2t, const float* __restrict__ b2,
    const int* __restrict__ bucket, const int* __restrict__ cnt,
    const float* __restrict__ pmaxArr, float* __restrict__ out)
{
  __shared__ __bf16 Xs[MT * XS_STR];     // gathered token rows, bf16   (~25 KB)
  __shared__ __bf16 Hs[MT * HS_STR];     // hidden activations, bf16    (~99 KB)
  __shared__ int   toks[MT];
  __shared__ float pms[MT];

  const int e    = blockIdx.y;
  const int tile = blockIdx.x;
  const int nTok = cnt[e];
  if (tile * MT >= nTok) return;          // uniform early exit (excess blocks)

  const int tid = threadIdx.x;
  if (tid < MT) {
    int idx = tile * MT + tid;
    int tk = (idx < nTok) ? bucket[(size_t)e * TOK + idx] : -1;
    toks[tid] = tk;
    pms[tid]  = (tk >= 0) ? pmaxArr[tk] : 0.f;
  }
  __syncthreads();

  // Gather X tile: MT rows x 256 bf16. Chunks of 32 elems (64B).
  for (int idx = tid; idx < MT * 8; idx += 256) {
    int row = idx >> 3;            // 0..MT-1
    int q   = idx & 7;             // eighth of a row
    int tk  = toks[row];
    uint4* dst = (uint4*)(Xs + row * XS_STR + q * 32);
    if (tk >= 0) {
      const uint4* src = (const uint4*)(xb + (size_t)tk * CDIM + q * 32);
#pragma unroll
      for (int i = 0; i < 4; ++i) dst[i] = src[i];
    } else {
#pragma unroll
      for (int i = 0; i < 4; ++i) dst[i] = make_uint4(0, 0, 0, 0);
    }
  }
  __syncthreads();

  const int wave  = tid >> 5;
  const int lane  = tid & 31;
  const int ln    = lane & 15;
  const int khalf = (lane < 16) ? 0 : 8;   // ISA A/B fragment k-offset

  // ---- Layer 1: h = relu(X[MTx256] * W1t^T -> [MTx1024]) ----
  // Each wave owns an nt stripe; B fragments are loaded ONCE and fed to
  // MTILES independent accumulators (weight reuse across M, deeper WMMA pipe).
  const __bf16* W1e = W1t + (size_t)e * HDIM * CDIM;
  for (int nt = wave; nt < HDIM / 16; nt += 8) {
    v8f acc[MTILES] = {};
    const __bf16* bptr = W1e + (size_t)(nt * 16 + ln) * CDIM + khalf;
#pragma unroll
    for (int k = 0; k < CDIM / 32; ++k) {
      v16bf b = load_frag(bptr + k * 32);
#pragma unroll
      for (int m = 0; m < MTILES; ++m) {
        v16bf a = load_frag(Xs + (m * 16 + ln) * XS_STR + khalf + k * 32);
        acc[m] = __builtin_amdgcn_wmma_f32_16x16x32_bf16(false, a, false, b,
                                                         (short)0, acc[m], false, false);
      }
    }
    float bias = b1[(size_t)e * HDIM + nt * 16 + ln];
#pragma unroll
    for (int m = 0; m < MTILES; ++m) {
#pragma unroll
      for (int r = 0; r < 8; ++r) {
        int mm = (lane < 16) ? r : (r + 8);
        float v = acc[m][r] + bias;
        v = v > 0.f ? v : 0.f;
        Hs[(m * 16 + mm) * HS_STR + nt * 16 + ln] = (__bf16)v;
      }
    }
  }
  __syncthreads();

  // ---- Layer 2: y = h[MTx1024] * W2t^T -> [MTx256]; scale by pmax; scatter ----
  const __bf16* W2e = W2t + (size_t)e * CDIM * HDIM;
  for (int nt = wave; nt < CDIM / 16; nt += 8) {
    v8f acc[MTILES] = {};
    const __bf16* bptr = W2e + (size_t)(nt * 16 + ln) * HDIM + khalf;
#pragma unroll 4
    for (int k = 0; k < HDIM / 32; ++k) {
      v16bf b = load_frag(bptr + k * 32);
#pragma unroll
      for (int m = 0; m < MTILES; ++m) {
        v16bf a = load_frag(Hs + (m * 16 + ln) * HS_STR + khalf + k * 32);
        acc[m] = __builtin_amdgcn_wmma_f32_16x16x32_bf16(false, a, false, b,
                                                         (short)0, acc[m], false, false);
      }
    }
    float bias = b2[(size_t)e * CDIM + nt * 16 + ln];
#pragma unroll
    for (int m = 0; m < MTILES; ++m) {
#pragma unroll
      for (int r = 0; r < 8; ++r) {
        int mm  = (lane < 16) ? r : (r + 8);
        int row = m * 16 + mm;
        int tk  = toks[row];
        if (tk >= 0)
          out[(size_t)tk * CDIM + nt * 16 + ln] = (acc[m][r] + bias) * pms[row];
      }
    }
  }
}

// ---------------- Host launcher ----------------
extern "C" void kernel_launch(void* const* d_in, const int* in_sizes, int n_in,
                              void* d_out, int out_size, void* d_ws, size_t ws_size,
                              hipStream_t stream) {
  const float* x   = (const float*)d_in[0];
  const float* Wsw = (const float*)d_in[1];
  const float* bsw = (const float*)d_in[2];
  const float* W1  = (const float*)d_in[3];
  const float* b1  = (const float*)d_in[4];
  const float* W2  = (const float*)d_in[5];
  const float* b2  = (const float*)d_in[6];
  float* out = (float*)d_out;

  // Workspace layout (256B aligned)
  char* ws = (char*)d_ws;
  size_t off = 0;
  auto take = [&](size_t bytes) { char* p = ws + off; off = (off + bytes + 255) & ~(size_t)255; return p; };
  __bf16* xb      = (__bf16*)take((size_t)TOK * CDIM * sizeof(__bf16));          // 32 MB
  __bf16* W1t     = (__bf16*)take((size_t)NEXP * HDIM * CDIM * sizeof(__bf16));  // 2 MB
  __bf16* W2t     = (__bf16*)take((size_t)NEXP * CDIM * HDIM * sizeof(__bf16));  // 2 MB
  float*  pmaxArr = (float*)take((size_t)TOK * sizeof(float));                   // 256 KB
  int*    bucket  = (int*)take((size_t)NEXP * TOK * sizeof(int));                // 1 MB
  int*    cnt     = (int*)take(NEXP * sizeof(int));

  hipMemsetAsync(cnt, 0, NEXP * sizeof(int), stream);

  prep_w1_kernel<<<(NEXP * HDIM * CDIM) / 256, 256, 0, stream>>>(W1, W1t);
  prep_w2_kernel<<<(NEXP * CDIM * HDIM) / 256, 256, 0, stream>>>(W2, W2t);

  router_kernel<<<TOK / 8, 256, 0, stream>>>(x, Wsw, bsw, xb, pmaxArr, bucket, cnt);

  dim3 grid((TOK + MT - 1) / MT, NEXP);
  moe_ffn_kernel<<<grid, 256, 0, stream>>>(xb, W1t, b1, W2t, b2,
                                           bucket, cnt, pmaxArr, out);
}
